// GQAttention_27702539059625
// MI455X (gfx1250) — matrix operand order repaired and checked by reading, not compiled, not run on hardware
//
#include <hip/hip_runtime.h>
#include <hip/hip_bf16.h>

// ---------------------------------------------------------------------------
// GQA attention for MI455X (gfx1250): all matmuls on v_wmma_f32_16x16x32_bf16
// cvt(bf16) -> Q/K/V projections (32x64-tile WMMA GEMM, Q pre-scaled by
// 1/sqrt(d)) -> flash attention (WMMA QK^T + online softmax with
// v_permlane16_b32 + v_pk_max_num_bf16 reduction, WMMA row-sum, WMMA PV)
// -> output projection (fp32 out).
// ---------------------------------------------------------------------------

typedef __bf16 bf16;
typedef __attribute__((ext_vector_type(16))) __bf16 v16bf;
typedef __attribute__((ext_vector_type(8)))  float  v8f;
typedef __attribute__((ext_vector_type(4)))  unsigned int u32x4;

#define SQ   4096
#define SKV  4096
#define DDIM 960
#define NH   15
#define NKV  5
#define HD   64
#define KVD  (NKV*HD)   /* 320 */

union FragU { v16bf v; u32x4 q[2]; };

// Pure compiler scheduling barrier (no instructions emitted): orders the
// type-punned LDS store/load pair that TBAA would otherwise allow to swap.
#define SCHED_FENCE() asm volatile("" ::: "memory")

// A-fragment (16x32, 16-bit): lane m = lane&15, sel = lane>>4.
__device__ __forceinline__ v16bf load_a(const bf16* row, int sel) {
  FragU f;
  f.q[0] = *(const u32x4*)(row + sel * 8);
  f.q[1] = *(const u32x4*)(row + 16 + sel * 8);
  return f.v;
}

// B-fragment (32x16, 16-bit): lane n = lane&15, sel = lane>>4.
__device__ __forceinline__ v16bf load_b(const bf16* col, int sel) {
  FragU f;
  f.q[0] = *(const u32x4*)(col + sel * 8);
  f.q[1] = *(const u32x4*)(col + 16 + sel * 8);
  return f.v;
}

__device__ __forceinline__ v8f wmma_bf(v16bf a, v16bf b, v8f c) {
  return __builtin_amdgcn_wmma_f32_16x16x32_bf16(false, a, false, b, (short)0, c, false, false);
}

__device__ __forceinline__ bf16 f2bf(float x) {          // RNE fp32 -> bf16
  unsigned u = __float_as_uint(x);
  u += 0x7FFFu + ((u >> 16) & 1u);
  unsigned short h = (unsigned short)(u >> 16);
  return __builtin_bit_cast(bf16, h);
}

__device__ __forceinline__ bf16 f2bf_trunc(float x) {    // truncating (cheap)
  unsigned short h = (unsigned short)(__float_as_uint(x) >> 16);
  return __builtin_bit_cast(bf16, h);
}

__device__ __forceinline__ v8f vzero8() {
  v8f z;
#pragma unroll
  for (int i = 0; i < 8; ++i) z[i] = 0.f;
  return z;
}

// raw v_max_num_f32 (skips fmaxf's sNaN canonicalization; WMMA outputs are
// never sNaN)
__device__ __forceinline__ float fmax_raw(float a, float b) {
  float d;
  asm("v_max_num_f32 %0, %1, %2" : "=v"(d) : "v"(a), "v"(b));
  return d;
}

// packed 2xbf16 max (VOP3P V_PK_MAX_NUM_BF16)
__device__ __forceinline__ unsigned pk_max_bf16(unsigned a, unsigned b) {
  unsigned d;
  asm("v_pk_max_num_bf16 %0, %1, %2" : "=v"(d) : "v"(a), "v"(b));
  return d;
}

// VALU cross-lane permute within 16-lane rows.
template <unsigned SEL0, unsigned SEL1>
__device__ __forceinline__ unsigned permlane16_u(unsigned src) {
  unsigned d;
  asm("v_permlane16_b32 %0, %1, %2, %3"
      : "=v"(d) : "v"(src), "s"(SEL0), "s"(SEL1));
  return d;
}

// One xor-reduction step over 16-lane groups: 4 permlanes + 4 packed maxes.
template <unsigned SEL0, unsigned SEL1>
__device__ __forceinline__ void pk_max_step(unsigned pk[4]) {
  unsigned t[4];
#pragma unroll
  for (int j = 0; j < 4; ++j) t[j] = permlane16_u<SEL0, SEL1>(pk[j]);
#pragma unroll
  for (int j = 0; j < 4; ++j) pk[j] = pk_max_bf16(pk[j], t[j]);
}

// ---------------------------------------------------------------------------
__global__ void __launch_bounds__(256) cvt_f32_bf16(const float* __restrict__ src,
                                                    bf16* __restrict__ dst, int n) {
  int i = blockIdx.x * 256 + threadIdx.x;
  if (i < n) dst[i] = f2bf(src[i]);
}

// W[k][n] (fp32, KxN) -> Wt[n][k] (bf16, NxK)
__global__ void __launch_bounds__(256) transp_cvt(const float* __restrict__ W,
                                                  bf16* __restrict__ Wt, int Kd, int Nd) {
  int i = blockIdx.x * 256 + threadIdx.x;
  if (i < Kd * Nd) {
    int k = i / Nd, n = i - k * Nd;
    Wt[(size_t)n * Kd + k] = f2bf(W[i]);
  }
}

// ---------------------------------------------------------------------------
// GEMM: C[M,N] = A[M,K] * B[K,N], B given as Bt[N][K] (bf16).
// One wave -> 32x64 output tile: 2 A-frags x 4 B-frags -> 8 WMMAs per k-step.
// OUT_MODE: 0 = bf16, 1 = bf16 transposed (out[n*M+m]), 2 = fp32,
//           3 = bf16 scaled by 0.125 (exact power-of-2, for Q).
template <int OUT_MODE>
__global__ void __launch_bounds__(256) gemm32x64(const bf16* __restrict__ A,
                                                 const bf16* __restrict__ Bt,
                                                 void* __restrict__ out,
                                                 int M, int N, int K) {
  int widx = threadIdx.x >> 5;
  int lane = threadIdx.x & 31;
  int wid = blockIdx.x * 8 + widx;
  int ntiles = N >> 6;
  if (wid >= (M >> 5) * ntiles) return;     // wave-uniform
  int m0 = (wid / ntiles) << 5;
  int n0 = (wid % ntiles) << 6;
  int lm = lane & 15, sel = lane >> 4;

  v8f acc[2][4];
#pragma unroll
  for (int s = 0; s < 2; ++s)
#pragma unroll
    for (int j = 0; j < 4; ++j) acc[s][j] = vzero8();

  const bf16* arow0 = A + (size_t)(m0 + lm) * K;
  const bf16* arow1 = arow0 + (size_t)16 * K;
  const bf16* brow  = Bt + (size_t)(n0 + lm) * K;
#pragma unroll 2
  for (int k0 = 0; k0 < K; k0 += 32) {
    v16bf af0 = load_a(arow0 + k0, sel);
    v16bf af1 = load_a(arow1 + k0, sel);
    v16bf b0 = load_b(brow + k0, sel);
    v16bf b1 = load_b(brow + (size_t)16 * K + k0, sel);
    v16bf b2 = load_b(brow + (size_t)32 * K + k0, sel);
    v16bf b3 = load_b(brow + (size_t)48 * K + k0, sel);
    acc[0][0] = wmma_bf(af0, b0, acc[0][0]);
    acc[0][1] = wmma_bf(af0, b1, acc[0][1]);
    acc[0][2] = wmma_bf(af0, b2, acc[0][2]);
    acc[0][3] = wmma_bf(af0, b3, acc[0][3]);
    acc[1][0] = wmma_bf(af1, b0, acc[1][0]);
    acc[1][1] = wmma_bf(af1, b1, acc[1][1]);
    acc[1][2] = wmma_bf(af1, b2, acc[1][2]);
    acc[1][3] = wmma_bf(af1, b3, acc[1][3]);
  }
#pragma unroll
  for (int s = 0; s < 2; ++s)
#pragma unroll
    for (int j = 0; j < 4; ++j)
#pragma unroll
      for (int r = 0; r < 8; ++r) {
        int m = m0 + 16 * s + r + 8 * sel;
        int n = n0 + 16 * j + lm;
        float v = acc[s][j][r];
        if (OUT_MODE == 0)      ((bf16*)out)[(size_t)m * N + n] = f2bf(v);
        else if (OUT_MODE == 1) ((bf16*)out)[(size_t)n * M + m] = f2bf(v);
        else if (OUT_MODE == 2) ((float*)out)[(size_t)m * N + n] = v;
        else                    ((bf16*)out)[(size_t)m * N + n] = f2bf(v * 0.125f);
      }
}

// ---------------------------------------------------------------------------
// Flash attention. One wave = 16 query rows x one head (d=64).
// Q: [SQ,960] (pre-scaled)  K: [SKV,320]  Vt: [320,SKV]  O: [SQ,960]  (bf16)
__global__ void __launch_bounds__(256) attn_fwd(const bf16* __restrict__ Q,
                                                const bf16* __restrict__ Kb,
                                                const bf16* __restrict__ Vt,
                                                bf16* __restrict__ O) {
  __shared__ __align__(16) bf16 pbuf[8 * 512];   // per-wave 16x32 P tile
  int widx = threadIdx.x >> 5;
  int lane = threadIdx.x & 31;
  int wid = blockIdx.x * 8 + widx;
  if (wid >= NH * (SQ / 16)) return;
  int head = wid >> 8;
  int q0 = (wid & 255) << 4;
  int kvh = head / 3;                  // KV_REPEAT = 3
  int lm = lane & 15, sel = lane >> 4;

  const bf16* qrow = Q + (size_t)(q0 + lm) * DDIM + head * HD;
  v16bf qf0 = load_a(qrow, sel);
  v16bf qf1 = load_a(qrow + 32, sel);

  v16bf onesf;                         // all-ones B matrix for row sums
#pragma unroll
  for (int i = 0; i < 16; ++i)
    onesf[i] = __builtin_bit_cast(bf16, (unsigned short)0x3F80);

  float mi[8], li[8];
  v8f o0 = vzero8(), o1 = vzero8(), o2 = vzero8(), o3 = vzero8();
#pragma unroll
  for (int r = 0; r < 8; ++r) { mi[r] = -1e30f; li[r] = 0.f; }
  bf16* pb = pbuf + widx * 512;

  const bf16* vrow = Vt + (size_t)(kvh * HD + lm) * SKV;

#pragma unroll 2
  for (int kv = 0; kv < SKV; kv += 32) {
    const bf16* k0p = Kb + (size_t)(kv + lm) * KVD + kvh * HD;
    const bf16* k1p = k0p + (size_t)16 * KVD;
    // hoist ALL fragment loads: 16 b128 loads in flight, waits stay partial
    v16bf kb0 = load_b(k0p, sel);
    v16bf kb1 = load_b(k0p + 32, sel);
    v16bf kb2 = load_b(k1p, sel);
    v16bf kb3 = load_b(k1p + 32, sel);
    const bf16* vbase = vrow + kv;
    v16bf vb0 = load_b(vbase, sel);
    v16bf vb1 = load_b(vbase + (size_t)16 * SKV, sel);
    v16bf vb2 = load_b(vbase + (size_t)32 * SKV, sel);
    v16bf vb3 = load_b(vbase + (size_t)48 * SKV, sel);
    if (kv + 32 < SKV) {
      __builtin_prefetch(k0p + 32 * KVD, 0, 0);
      __builtin_prefetch(k1p + 32 * KVD, 0, 0);
    }
    v8f s0 = vzero8(), s1 = vzero8();
    s0 = wmma_bf(qf0, kb0, s0);
    s0 = wmma_bf(qf1, kb1, s0);
    s1 = wmma_bf(qf0, kb2, s1);
    s1 = wmma_bf(qf1, kb3, s1);

    // Q was pre-scaled by 1/8, so s0/s1 are final logits.
    float p0[8], p1[8], mx[8];
#pragma unroll
    for (int r = 0; r < 8; ++r) {
      p0[r] = s0[r];
      p1[r] = s1[r];
      mx[r] = fmax_raw(p0[r], p1[r]);
    }
    // row-max across each 16-lane group: rows packed pairwise as bf16,
    // 4 xor steps of (4x v_permlane16_b32 + 4x v_pk_max_num_bf16) — pure VALU.
    unsigned pk[4];
#pragma unroll
    for (int j = 0; j < 4; ++j)
      pk[j] = (__float_as_uint(mx[2 * j]) >> 16) |
              (__float_as_uint(mx[2 * j + 1]) & 0xFFFF0000u);
    pk_max_step<0x67452301u, 0xEFCDAB89u>(pk);   // xor 1
    pk_max_step<0x54761032u, 0xDCFE98BAu>(pk);   // xor 2
    pk_max_step<0x32107654u, 0xBA98FEDCu>(pk);   // xor 4
    pk_max_step<0xFEDCBA98u, 0x76543210u>(pk);   // xor 8
#pragma unroll
    for (int j = 0; j < 4; ++j) {
      mx[2 * j]     = __uint_as_float(pk[j] << 16);
      mx[2 * j + 1] = __uint_as_float(pk[j] & 0xFFFF0000u);
    }
#pragma unroll
    for (int r = 0; r < 8; ++r) {
      float mnew = fmax_raw(mi[r], mx[r]);
      float alpha = __expf(mi[r] - mnew);
      mi[r] = mnew;
      p0[r] = __expf(p0[r] - mnew);
      p1[r] = __expf(p1[r] - mnew);
      o0[r] *= alpha; o1[r] *= alpha; o2[r] *= alpha; o3[r] *= alpha;
      li[r] *= alpha;
    }
    // C-frag -> row-major bf16 P tile in per-wave LDS (ds ops; the empty-asm
    // fences only order the compiler — same-wave DS pipeline is in-order).
    SCHED_FENCE();
#pragma unroll
    for (int r = 0; r < 8; ++r) {
      int m = r + 8 * sel;
      pb[m * 32 + lm]      = f2bf_trunc(p0[r]);
      pb[m * 32 + 16 + lm] = f2bf_trunc(p1[r]);
    }
    SCHED_FENCE();
    v16bf pf = load_a(pb + lm * 32, sel);
    // row sums of the SAME bf16 P via WMMA against ones
    v8f rsum = wmma_bf(pf, onesf, vzero8());
    o0 = wmma_bf(pf, vb0, o0);
    o1 = wmma_bf(pf, vb1, o1);
    o2 = wmma_bf(pf, vb2, o2);
    o3 = wmma_bf(pf, vb3, o3);
#pragma unroll
    for (int r = 0; r < 8; ++r) li[r] += rsum[r];
  }
#pragma unroll
  for (int r = 0; r < 8; ++r) {
    float inv = 1.f / li[r];
    int m = q0 + r + 8 * sel;
    bf16* orow = O + (size_t)m * DDIM + head * HD;
    orow[lm]      = f2bf(o0[r] * inv);
    orow[16 + lm] = f2bf(o1[r] * inv);
    orow[32 + lm] = f2bf(o2[r] * inv);
    orow[48 + lm] = f2bf(o3[r] * inv);
  }
}

// ---------------------------------------------------------------------------
extern "C" void kernel_launch(void* const* d_in, const int* in_sizes, int n_in,
                              void* d_out, int out_size, void* d_ws, size_t ws_size,
                              hipStream_t stream) {
  const float* q_in  = (const float*)d_in[0];
  const float* kv_in = (const float*)d_in[1];
  const float* Wq = (const float*)d_in[2];
  const float* Wk = (const float*)d_in[3];
  const float* Wv = (const float*)d_in[4];
  const float* Wo = (const float*)d_in[5];

  char* ws = (char*)d_ws;
  size_t off = 0;
  auto alloc = [&](size_t elems) {
    void* p = ws + off;
    off += (elems * sizeof(bf16) + 255) & ~(size_t)255;
    return (bf16*)p;
  };
  bf16* qbf  = alloc((size_t)SQ * DDIM);
  bf16* kvbf = alloc((size_t)SKV * DDIM);
  bf16* wqt  = alloc((size_t)DDIM * DDIM);
  bf16* wkt  = alloc((size_t)KVD * DDIM);
  bf16* wvt  = alloc((size_t)KVD * DDIM);
  bf16* wot  = alloc((size_t)DDIM * DDIM);
  bf16* Qb   = alloc((size_t)SQ * DDIM);     // [4096, 960]  (pre-scaled)
  bf16* Kb   = alloc((size_t)SKV * KVD);     // [4096, 320]
  bf16* Vtb  = alloc((size_t)KVD * SKV);     // [320, 4096]
  bf16* Ob   = alloc((size_t)SQ * DDIM);
  (void)ws_size; (void)n_in; (void)in_sizes; (void)out_size;

  const int nAct = SQ * DDIM;
  cvt_f32_bf16<<<(nAct + 255) / 256, 256, 0, stream>>>(q_in,  qbf,  nAct);
  cvt_f32_bf16<<<(nAct + 255) / 256, 256, 0, stream>>>(kv_in, kvbf, nAct);
  transp_cvt<<<(DDIM * DDIM + 255) / 256, 256, 0, stream>>>(Wq, wqt, DDIM, DDIM);
  transp_cvt<<<(DDIM * KVD + 255) / 256, 256, 0, stream>>>(Wk, wkt, DDIM, KVD);
  transp_cvt<<<(DDIM * KVD + 255) / 256, 256, 0, stream>>>(Wv, wvt, DDIM, KVD);
  transp_cvt<<<(DDIM * DDIM + 255) / 256, 256, 0, stream>>>(Wo, wot, DDIM, DDIM);

  // projections: wave = 32x64 tile, 8 waves/block. Q pre-scaled by 1/8.
  gemm32x64<3><<<(SQ / 32) * (DDIM / 64) / 8, 256, 0, stream>>>(qbf,  wqt, Qb,  SQ,  DDIM, DDIM);
  gemm32x64<0><<<(SKV / 32) * (KVD / 64) / 8, 256, 0, stream>>>(kvbf, wkt, Kb,  SKV, KVD,  DDIM);
  gemm32x64<1><<<(SKV / 32) * (KVD / 64) / 8, 256, 0, stream>>>(kvbf, wvt, Vtb, SKV, KVD,  DDIM);

  // flash attention: 15 heads x 256 q-tiles = 3840 wave tasks
  attn_fwd<<<NH * (SQ / 16) / 8, 256, 0, stream>>>(Qb, Kb, Vtb, Ob);

  // output projection -> fp32 d_out
  gemm32x64<2><<<(SQ / 32) * (DDIM / 64) / 8, 256, 0, stream>>>(Ob, wot, d_out, SQ, DDIM, DDIM);
}
